// BreakthroughSNN_11416023073064
// MI455X (gfx1250) — matrix operand order repaired and checked by reading, not compiled, not run on hardware
//
#include <hip/hip_runtime.h>
#include <hip/hip_bf16.h>

// Problem constants (from reference)
#define V_   50257
#define D_   512
#define DS_  256
#define L_   2
#define T_   2
#define B_   4
#define S_   256
#define BETA 0.95f
#define THR  1.0f

typedef __attribute__((ext_vector_type(16))) __bf16 bf16x16;
typedef __attribute__((ext_vector_type(8)))  float  f32x8;

// ---------------------------------------------------------------------------
// WMMA helper: D = A(16x32 bf16) x B(32x16 bf16) + C(16x16 f32)
// Fragment element maps (ISA 7.12.2):
//   A: lane m=lane&15, h=lane>>4; elem i -> K = 8h+i (i<8) else 16+8h+(i-8)
//   B: lane n=lane&15, h=lane>>4; elem i -> K = 8h+i (i<8) else 16+8h+(i-8)
//   C: elem r -> row r+8h, col lane&15
// ---------------------------------------------------------------------------
__device__ __forceinline__ f32x8 wmma_bf16(bf16x16 a, bf16x16 b, f32x8 c) {
  return __builtin_amdgcn_wmma_f32_16x16x32_bf16(
      /*neg_a=*/false, a, /*neg_b=*/false, b,
      /*c_mod=*/(short)0, c, /*reuse_a=*/false, /*reuse_b=*/false);
}

// A-fragment elem i -> K offset within the 32-wide k-tile
__device__ __forceinline__ int a_koff(int h, int i) {
  return i < 8 ? 8 * h + i : 8 + 8 * h + i;
}
// Inverse map: k-offset r (0..31) -> (h, i)
__device__ __forceinline__ void a_inv(int r, int& h, int& i) {
  h = (r >> 3) & 1;
  i = (r & 7) + (r >= 16 ? 8 : 0);
}

// A fragment gathered through embedding table (f32 -> bf16)
__device__ __forceinline__ bf16x16 frag_A_embed(const float* emb, const int* ids,
                                                int m0, int k0, int lane) {
  int m = m0 + (lane & 15), h = lane >> 4;
  const float* row = emb + (size_t)ids[m] * D_;
  bf16x16 a;
#pragma unroll
  for (int i = 0; i < 16; ++i) a[i] = (__bf16)row[k0 + a_koff(h, i)];
  return a;
}

// B fragment from pre-swizzled bf16 weights: layout [nt][kt][lane][16]
// -> one 32-byte vector load per lane (2x b128).
__device__ __forceinline__ bf16x16 frag_B_swz(const __bf16* base, int nt, int kt,
                                              int KT, int lane) {
  return *(const bf16x16*)(base + ((((size_t)nt * KT + kt) * 32 + lane) << 4));
}

// B fragment from f32 row-major [K x N] with column guard (f32 -> bf16)
__device__ __forceinline__ bf16x16 frag_B_f32(const float* Bm, int ldb, int k0,
                                              int n0, int lane, int nmax) {
  int n = n0 + (lane & 15), h = lane >> 4;
  bf16x16 b;
#pragma unroll
  for (int i = 0; i < 16; ++i) {
    float v = (n < nmax) ? Bm[(size_t)(k0 + (i < 8 ? 8 * h + i : 8 + 8 * h + i)) * ldb + n] : 0.0f;
    b[i] = (__bf16)v;
  }
  return b;
}

__device__ __forceinline__ float wave_sum(float v) {
#pragma unroll
  for (int off = 16; off > 0; off >>= 1) v += __shfl_xor(v, off, 32);
  return v;
}

// ---------------------------------------------------------------------------
// CDNA5 async global -> LDS copy (ASYNCcnt path). Each active lane moves 16B.
// ---------------------------------------------------------------------------
__device__ __forceinline__ void async_copy_b128(const void* gsrc, void* lds_dst) {
  unsigned ldsoff = (unsigned)(size_t)lds_dst;
  unsigned long long ga = (unsigned long long)(size_t)gsrc;
  asm volatile("global_load_async_to_lds_b128 %0, %1, off"
               :: "v"(ldsoff), "v"(ga) : "memory");
}

__device__ __forceinline__ void wait_async0() {
#if __has_builtin(__builtin_amdgcn_s_wait_asynccnt)
  __builtin_amdgcn_s_wait_asynccnt(0);
#else
  asm volatile("s_wait_asynccnt 0x0" ::: "memory");
#endif
}

// ---------------------------------------------------------------------------
// Kernel 0: f32 [layers][K][N] -> bf16 fragment-swizzled [layer][nt][kt][lane][16]
// ---------------------------------------------------------------------------
__global__ void swz_weight_bf16(const float* __restrict__ src,
                                __bf16* __restrict__ dst,
                                int K, int N, int layers) {
  int total = layers * K * N;
  int e = blockIdx.x * blockDim.x + threadIdx.x;
  if (e >= total) return;
  int KT = K >> 5, NT = N >> 4;
  int i    = e & 15;
  int lane = (e >> 4) & 31;
  int kt   = (e >> 9) % KT;
  int rest = (e >> 9) / KT;
  int nt   = rest % NT;
  int lyr  = rest / NT;
  int h = lane >> 4;
  int k = kt * 32 + (i < 8 ? 8 * h + i : 8 + 8 * h + i);  // B-fragment map: K = 8h+i (i<8) else 16+8h+(i-8)
  int n = nt * 16 + (lane & 15);
  dst[e] = (__bf16)src[(size_t)lyr * K * N + (size_t)k * N + n];
}

// ---------------------------------------------------------------------------
// Kernel 1: seq[b*S+s][d] = emb[ids[b*S+s]] @ enc_w + enc_b
// M=1024, N=512, K=512. One wave per 16x16 output tile.
// ---------------------------------------------------------------------------
__global__ void __launch_bounds__(256)
snn_embed_encode(const int* __restrict__ ids, const float* __restrict__ emb,
                 const __bf16* __restrict__ encw, const float* __restrict__ enc_b,
                 float* __restrict__ seq) {
  int lane = threadIdx.x & 31, w = threadIdx.x >> 5;
  int tile = blockIdx.x * 8 + w;       // 2048 tiles total
  int m0 = (tile >> 5) * 16;           // 64 M tiles
  int nt = tile & 31;                  // 32 N tiles
  f32x8 c = {};
#pragma unroll
  for (int kt = 0; kt < D_ / 32; ++kt) {
    bf16x16 a = frag_A_embed(emb, ids, m0, kt * 32, lane);
    bf16x16 b = frag_B_swz(encw, nt, kt, D_ / 32, lane);
    c = wmma_bf16(a, b, c);
  }
  int n = nt * 16 + (lane & 15), h = lane >> 4;
  float bias = enc_b[n];
#pragma unroll
  for (int r = 0; r < 8; ++r) {
    int m = m0 + r + 8 * h;
    seq[m * D_ + n] = c[r] + bias;
  }
}

// ---------------------------------------------------------------------------
// Kernel 2: sequential T x S scan on a single WGP. LayerNorm writes directly
// into a bf16 A-fragment-swizzled LDS buffer (s_xa[kt][lane][16]); every wave
// then reads its A fragment as one 32B vector (2x ds_load_b128). B fragments
// stream from pre-swizzled weights as 2x global_load_b128 (L2-resident).
// ---------------------------------------------------------------------------
__global__ void __launch_bounds__(1024)
snn_recurrent(const float* __restrict__ seq, const __bf16* __restrict__ genw,
              const __bf16* __restrict__ infw, const float* __restrict__ gen_b,
              const float* __restrict__ inf_b, const float* __restrict__ ns_g,
              const float* __restrict__ ns_b, const float* __restrict__ ne_g,
              const float* __restrict__ ne_b, const float* __restrict__ err_s,
              __bf16* __restrict__ A_bf) {
  __shared__ float s_st[L_][B_][DS_];                 //  8 KB  states
  __shared__ float s_gm[L_][B_][D_];                  // 16 KB  generator membrane
  __shared__ float s_im[L_][B_][DS_];                 //  8 KB  inference membrane
  __shared__ float s_bu[B_][D_];                      //  8 KB  bottom-up / error
  __shared__ __align__(32) __bf16 s_xa[16][32][16];   // 16 KB  A fragments (swz)

  int tid = threadIdx.x, lane = tid & 31, w = tid >> 5;

  for (int i = tid; i < L_ * B_ * DS_; i += 1024) {
    ((float*)s_st)[i] = 0.0f;
    ((float*)s_im)[i] = 0.0f;
  }
  for (int i = tid; i < L_ * B_ * D_; i += 1024) ((float*)s_gm)[i] = 0.0f;
  for (int i = tid; i < 16 * 32 * 16; i += 1024) ((__bf16*)s_xa)[i] = (__bf16)0.0f;
  __syncthreads();

  for (int t = 0; t < T_; ++t) {
    for (int s = 0; s < S_; ++s) {
      // bu = seq_t[s]  (seq stored [b*S+s][d])
      for (int i = tid; i < B_ * D_; i += 1024) {
        int b = i >> 9, d = i & 511;
        s_bu[b][d] = seq[(b * S_ + s) * D_ + d];
      }
      __syncthreads();

      for (int j = 0; j < L_; ++j) {
        // ---- LN over states[j] (rows B_, width DS_) -> s_xa, waves 0..3 ----
        if (w < B_) {
          float v[8], sum = 0.f, sq = 0.f;
#pragma unroll
          for (int i = 0; i < 8; ++i) {
            v[i] = s_st[j][w][i * 32 + lane];
            sum += v[i]; sq += v[i] * v[i];
          }
          sum = wave_sum(sum); sq = wave_sum(sq);
          float mu = sum * (1.0f / DS_);
          float var = sq * (1.0f / DS_) - mu * mu;
          float rs = rsqrtf(var + 1e-5f);
#pragma unroll
          for (int i = 0; i < 8; ++i) {
            int k = i * 32 + lane;
            float y = (v[i] - mu) * rs * ns_g[j * DS_ + k] + ns_b[j * DS_ + k];
            int hh, ii; a_inv(k & 31, hh, ii);
            s_xa[k >> 5][w + 16 * hh][ii] = (__bf16)y;
          }
        }
        __syncthreads();

        // ---- GEMM1: pred_cur[4,512] = LN(states) @ gen_w[j]; LIF; err ----
        {
          f32x8 c = {};
          const __bf16* Bw = genw + j * DS_ * D_;
#pragma unroll
          for (int kt = 0; kt < DS_ / 32; ++kt) {
            bf16x16 a = *(const bf16x16*)&s_xa[kt][lane][0];
            bf16x16 b = frag_B_swz(Bw, w, kt, DS_ / 32, lane);
            c = wmma_bf16(a, b, c);
          }
          int n = w * 16 + (lane & 15);
          if (lane < 16) {               // rows 0..3 live in c[0..3] of lanes 0-15
            float es = err_s[j];
#pragma unroll
            for (int r = 0; r < B_; ++r) {
              float cur = c[r] + gen_b[j * D_ + n];
              float gm = BETA * s_gm[j][r][n] + cur;
              float spk = (gm - THR) > 0.0f ? 1.0f : 0.0f;
              s_gm[j][r][n] = gm - spk * THR;
              s_bu[r][n] = (s_bu[r][n] - spk) * es;   // err -> becomes next bu
            }
          }
        }
        __syncthreads();

        // ---- LN over err (rows B_, width D_) -> s_xa, waves 0..3 ----
        if (w < B_) {
          float v[16], sum = 0.f, sq = 0.f;
#pragma unroll
          for (int i = 0; i < 16; ++i) {
            v[i] = s_bu[w][i * 32 + lane];
            sum += v[i]; sq += v[i] * v[i];
          }
          sum = wave_sum(sum); sq = wave_sum(sq);
          float mu = sum * (1.0f / D_);
          float var = sq * (1.0f / D_) - mu * mu;
          float rs = rsqrtf(var + 1e-5f);
#pragma unroll
          for (int i = 0; i < 16; ++i) {
            int k = i * 32 + lane;
            float y = (v[i] - mu) * rs * ne_g[j * D_ + k] + ne_b[j * D_ + k];
            int hh, ii; a_inv(k & 31, hh, ii);
            s_xa[k >> 5][w + 16 * hh][ii] = (__bf16)y;
          }
        }
        __syncthreads();

        // ---- GEMM2: inf_cur[4,256] = LN(err) @ inf_w[j]; LIF; state += ----
        if (w < 16) {                    // 16 waves x 16 = 256 = DS_
          f32x8 c = {};
          const __bf16* Bw = infw + j * D_ * DS_;
#pragma unroll
          for (int kt = 0; kt < D_ / 32; ++kt) {
            bf16x16 a = *(const bf16x16*)&s_xa[kt][lane][0];
            bf16x16 b = frag_B_swz(Bw, w, kt, D_ / 32, lane);
            c = wmma_bf16(a, b, c);
          }
          int n = w * 16 + (lane & 15);
          if (lane < 16) {
#pragma unroll
            for (int r = 0; r < B_; ++r) {
              float cur = c[r] + inf_b[j * DS_ + n];
              float im = BETA * s_im[j][r][n] + cur;
              float spk = (im - THR) > 0.0f ? 1.0f : 0.0f;
              s_im[j][r][n] = im - spk * THR;
              s_st[j][r][n] += spk;
            }
          }
        }
        __syncthreads();
      } // layers

      // final-timestep states -> bf16 A matrix, A-fragment swizzled per m-tile:
      // layout [mtile][kt][lane][16], m = b*S + s, k = j*DS + n
      if (t == T_ - 1) {
        for (int idx = tid; idx < B_ * L_ * DS_; idx += 1024) {
          int b = idx >> 9, k = idx & 511;
          int j = k >> 8, n = k & 255;
          int m = b * S_ + s;
          int hh, ii; a_inv(k & 31, hh, ii);
          size_t off = ((((size_t)(m >> 4) * 16 + (k >> 5)) * 32)
                        + (m & 15) + 16 * hh) * 16 + ii;
          A_bf[off] = (__bf16)s_st[j][b][n];
        }
      }
    } // s
  }   // t
}

// ---------------------------------------------------------------------------
// Kernel 3: out[m,:] = A @ out_w + out_b   (M=1024, N=50257, K=512)
// A_bf is fragment-swizzled: [mtile][kt][lane][16]. Panels of 2 m-tiles
// (32 KB) are double-buffered into LDS with global_load_async_to_lds_b128;
// fragment reads are single 32B vectors (2x ds_load_b128). Each wave keeps
// all 16 B fragments of its 16-column strip in VGPRs (out_w read once).
// ---------------------------------------------------------------------------
__global__ void __launch_bounds__(512)
snn_logits(const __bf16* __restrict__ A_bf, const float* __restrict__ out_w,
           const float* __restrict__ out_b, float* __restrict__ out) {
  __shared__ __align__(32) __bf16 sA[2][2 * 16 * 32 * 16];  // 2 x 32 KB panels
  int tid = threadIdx.x, lane = tid & 31, w = tid >> 5;
  int n0 = (blockIdx.x * 16 + w) * 16;
  bool active = n0 < V_;

  if (active) __builtin_prefetch(out_w + n0, 0, 0);   // global_prefetch_b8

  // Preload all B fragments for this wave's 16-column strip (K = 512).
  bf16x16 bfr[16];
  if (active) {
#pragma unroll
    for (int kk = 0; kk < 16; ++kk)
      bfr[kk] = frag_B_f32(out_w, V_, kk * 32, n0, lane, V_);
  }

  // Stage panel 0, then pipeline: issue async for mc+1, compute mc.
  for (int i = tid; i < (2 * 16 * 512) / 8; i += 512)
    async_copy_b128(A_bf + i * 8, &sA[0][i * 8]);
  wait_async0();
  __syncthreads();

  for (int mc = 0; mc < 32; ++mc) {    // 32 panels of 2 m-tiles (32 M rows)
    int buf = mc & 1;

    if (mc + 1 < 32) {                 // overlap next panel DMA with compute
      const __bf16* src = A_bf + (size_t)(mc + 1) * 2 * 16 * 512;
      for (int i = tid; i < (2 * 16 * 512) / 8; i += 512)
        async_copy_b128(src + i * 8, &sA[buf ^ 1][i * 8]);
    }

    if (active) {
#pragma unroll
      for (int mt = 0; mt < 2; ++mt) {
        f32x8 c = {};
#pragma unroll
        for (int kt = 0; kt < 16; ++kt) {
          bf16x16 a = *(const bf16x16*)&sA[buf][((mt * 16 + kt) * 32 + lane) * 16];
          c = wmma_bf16(a, bfr[kt], c);
        }
        int n = n0 + (lane & 15), h = lane >> 4;
        if (n < V_) {
          float bias = out_b[n];
#pragma unroll
          for (int r = 0; r < 8; ++r) {
            int m = mc * 32 + mt * 16 + r + 8 * h;
            out[(size_t)m * V_ + n] = c[r] + bias;
          }
        }
      }
    }

    wait_async0();                     // this wave's panel mc+1 copies done
    __syncthreads();                   // all waves done staging + computing
  }
}

// ---------------------------------------------------------------------------
// Launch
// ---------------------------------------------------------------------------
extern "C" void kernel_launch(void* const* d_in, const int* in_sizes, int n_in,
                              void* d_out, int out_size, void* d_ws, size_t ws_size,
                              hipStream_t stream) {
  const int*   ids   = (const int*)  d_in[0];
  const float* emb   = (const float*)d_in[1];
  const float* enc_w = (const float*)d_in[2];
  const float* enc_b = (const float*)d_in[3];
  const float* gen_w = (const float*)d_in[4];
  const float* gen_b = (const float*)d_in[5];
  const float* inf_w = (const float*)d_in[6];
  const float* inf_b = (const float*)d_in[7];
  const float* ns_g  = (const float*)d_in[8];
  const float* ns_b  = (const float*)d_in[9];
  const float* ne_g  = (const float*)d_in[10];
  const float* ne_b  = (const float*)d_in[11];
  const float* err_s = (const float*)d_in[12];
  const float* out_w = (const float*)d_in[13];
  const float* out_b = (const float*)d_in[14];
  float* out = (float*)d_out;

  char* ws = (char*)d_ws;
  float*  seq  = (float*) (ws);                                  // 2 MB
  __bf16* A_bf = (__bf16*)(ws + (size_t)(2u << 20));             // 1 MB
  __bf16* encw = (__bf16*)(ws + (size_t)(4u << 20));             // 512 KB
  __bf16* genw = (__bf16*)(ws + (size_t)(4u << 20) + (512u << 10));
  __bf16* infw = (__bf16*)(ws + (size_t)(5u << 20));             // 512 KB

  const int NW = L_ * DS_ * D_;  // 262144 elements each
  swz_weight_bf16<<<(D_ * D_ + 255) / 256, 256, 0, stream>>>(enc_w, encw, D_, D_, 1);
  swz_weight_bf16<<<(NW + 255) / 256, 256, 0, stream>>>(gen_w, genw, DS_, D_, L_);
  swz_weight_bf16<<<(NW + 255) / 256, 256, 0, stream>>>(inf_w, infw, D_, DS_, L_);

  snn_embed_encode<<<256, 256, 0, stream>>>(ids, emb, encw, enc_b, seq);
  snn_recurrent<<<1, 1024, 0, stream>>>(seq, genw, infw, gen_b, inf_b,
                                        ns_g, ns_b, ne_g, ne_b, err_s, A_bf);
  snn_logits<<<(V_ + 255) / 256, 512, 0, stream>>>(A_bf, out_w, out_b, out);
}
